// ImplicitEGNN_44796508897965
// MI455X (gfx1250) — matrix operand order repaired and checked by reading, not compiled, not run on hardware
//
#include <hip/hip_runtime.h>

typedef _Float16 v16h __attribute__((ext_vector_type(16)));
typedef _Float16 v8h  __attribute__((ext_vector_type(8)));
typedef _Float16 h4   __attribute__((ext_vector_type(4)));
typedef float    v8f  __attribute__((ext_vector_type(8)));

#define LDA 136   // f16 LDS row stride for 128-wide tiles (16B aligned, bank-skewed)
#define LDG 72    // f16 LDS row stride for 64-wide tiles
#define LDM 68    // f32 LDS row stride for 64-wide m tile

#define WMMA_F16(A,B,C) __builtin_amdgcn_wmma_f32_16x16x32_f16(false,(A),false,(B),(short)0,(C),false,false)

__device__ __forceinline__ float gelu_t(float x){
  const float c = 0.7978845608028654f;            // sqrt(2/pi)
  float x3 = x*x*x;
  return 0.5f * x * (1.0f + tanhf(c*(x + 0.044715f*x3)));
}

__device__ __forceinline__ void atom_add(float* p, float v){
  unsafeAtomicAdd(p, v);   // native global_atomic_add_f32 on CDNA
}

// A fragment (16x32 f16) from LDS, row-major tile with given stride.
// lane holds row m=lane&15; elems 0..7 -> K=kb..kb+7, elems 8..15 -> K=kb+16..kb+23
__device__ __forceinline__ v16h load_a_frag(const _Float16* base, int stride, int row, int k0, int kb){
  const _Float16* p = base + row*stride + k0 + kb;
  v16h a;
  ((v8h*)&a)[0] = *(const v8h*)(p);
  ((v8h*)&a)[1] = *(const v8h*)(p + 16);
  return a;
}

// ---------------------------------------------------------------- utility
__global__ __launch_bounds__(256) void zero_kernel(float* __restrict__ p, long n){
  long i = (long)blockIdx.x*blockDim.x + threadIdx.x;
  long s = (long)gridDim.x*blockDim.x;
  for (; i < n; i += s) p[i] = 0.0f;
}

__global__ __launch_bounds__(256) void count_kernel(const int* __restrict__ dst, float* __restrict__ cnt, int E){
  int i = blockIdx.x*blockDim.x + threadIdx.x;
  if (i < E) atom_add(&cnt[dst[i]], 1.0f);   // adds of 1.0f are exact -> order-independent
}

// Pack row-major f32 weight [K x Nc] (leading dim ld) into WMMA-B fragments:
// out[((kt*(Nc/16)+nt)*32 + lane)*16 + i] = W[kt*32 + (lane>>4)*8 + (i&7) + (i>>3)*16][nt*16 + (lane&15)]
__global__ __launch_bounds__(256) void pack_b_kernel(const float* __restrict__ W, int K, int Nc, int ld,
                                                     _Float16* __restrict__ out){
  int total = (K/32)*(Nc/16)*512;
  for (int idx = blockIdx.x*blockDim.x + threadIdx.x; idx < total; idx += gridDim.x*blockDim.x){
    int i    = idx & 15;
    int lane = (idx >> 4) & 31;
    int t    = idx >> 9;
    int ntiles = Nc/16;
    int nt = t % ntiles, kt = t / ntiles;
    int k = kt*32 + ((lane>>4)*8) + (i&7) + ((i>>3)*16);
    int n = nt*16 + (lane&15);
    out[idx] = (_Float16)W[(size_t)k*ld + n];
  }
}

// ---------------------------------------------------------------- LayerNorm
// mode 0: out = LN(a) ; mode 1: out = gelu(LN(a+b)) ; mode 2: out = LN(gelu(a+b))
__global__ __launch_bounds__(256) void ln_kernel(const float* __restrict__ a, const float* __restrict__ b,
                                                 float* __restrict__ out, int N, int mode){
  int row  = blockIdx.x*8 + (threadIdx.x >> 5);
  int lane = threadIdx.x & 31;
  if (row >= N) return;
  const float* pa = a + (size_t)row*128;
  float v[4];
  #pragma unroll
  for (int i=0;i<4;++i){
    float t = pa[lane + 32*i];
    if (mode != 0) t += b[(size_t)row*128 + lane + 32*i];
    if (mode == 2) t = gelu_t(t);
    v[i] = t;
  }
  float s = v[0]+v[1]+v[2]+v[3];
  #pragma unroll
  for (int m=16;m>=1;m>>=1) s += __shfl_xor(s, m, 32);
  float mu = s * (1.0f/128.0f);
  float q = 0.f;
  #pragma unroll
  for (int i=0;i<4;++i){ float d = v[i]-mu; q += d*d; }
  #pragma unroll
  for (int m=16;m>=1;m>>=1) q += __shfl_xor(q, m, 32);
  float rstd = rsqrtf(q*(1.0f/128.0f) + 1e-5f);
  #pragma unroll
  for (int i=0;i<4;++i){
    float y = (v[i]-mu)*rstd;
    if (mode == 1) y = gelu_t(y);
    out[(size_t)row*128 + lane + 32*i] = y;
  }
}

// ---------------------------------------------------------------- BatchNorm (N x 3), deterministic 2-stage
__global__ __launch_bounds__(256) void bn_reduce_kernel(const float* __restrict__ a, const float* __restrict__ b,
                                                        float* __restrict__ partial, int N){
  __shared__ float sh[8][6];
  float s[3] = {0,0,0}, q[3] = {0,0,0};
  for (int r = blockIdx.x*blockDim.x + threadIdx.x; r < N; r += gridDim.x*blockDim.x){
    #pragma unroll
    for (int k=0;k<3;++k){
      float t = a[r*3+k] + b[r*3+k];
      s[k] += t; q[k] += t*t;
    }
  }
  #pragma unroll
  for (int k=0;k<3;++k){
    #pragma unroll
    for (int m=16;m>=1;m>>=1){ s[k] += __shfl_xor(s[k], m, 32); q[k] += __shfl_xor(q[k], m, 32); }
  }
  int wv = threadIdx.x >> 5, lane = threadIdx.x & 31;
  if (lane == 0){
    #pragma unroll
    for (int k=0;k<3;++k){ sh[wv][k] = s[k]; sh[wv][3+k] = q[k]; }
  }
  __syncthreads();
  if (threadIdx.x < 6){
    float t = 0.f;
    #pragma unroll
    for (int w=0; w<8; ++w) t += sh[w][threadIdx.x];
    partial[blockIdx.x*6 + threadIdx.x] = t;
  }
}

__global__ void bn_final_kernel(const float* __restrict__ partial, int nblk, float* __restrict__ stats){
  int k = threadIdx.x;
  if (k < 6){
    float s = 0.f;
    for (int b = 0; b < nblk; ++b) s += partial[b*6 + k];   // fixed order -> deterministic
    stats[k] = s;
  }
}

__global__ __launch_bounds__(256) void bn_apply_kernel(const float* __restrict__ a, const float* __restrict__ b,
                                                       const float* __restrict__ stats, float* __restrict__ out, int N){
  int i = blockIdx.x*blockDim.x + threadIdx.x;
  if (i < N*3){
    int k = i % 3;
    float invN = 1.0f/(float)N;
    float mu  = stats[k]*invN;
    float var = stats[3+k]*invN - mu*mu;
    float t = a[i] + b[i];
    out[i] = (t - mu) * rsqrtf(var + 1e-5f);
  }
}

// ---------------------------------------------------------------- GEMM 128x128 (+bias, gelu) for inject
__global__ __launch_bounds__(256) void gemm128_gelu_kernel(const float* __restrict__ in,
    const _Float16* __restrict__ Wp, const float* __restrict__ bias, float* __restrict__ out, int N){
  __shared__ _Float16 sX[64*LDA];
  const int tid = threadIdx.x;
  const int n0  = blockIdx.x*64;
  {
    int r = tid >> 2, c0 = (tid & 3) * 32;
    int row = n0 + r; if (row >= N) row = N-1;
    const float4* px = (const float4*)(in + (size_t)row*128 + c0);
    _Float16* q = &sX[r*LDA + c0];
    #pragma unroll
    for (int k=0;k<8;++k){
      float4 v = px[k];
      h4 h = { (_Float16)v.x, (_Float16)v.y, (_Float16)v.z, (_Float16)v.w };
      *(h4*)(q + 4*k) = h;
    }
  }
  __syncthreads();
  const int lane = tid & 31, wv = tid >> 5;
  const int mrow = lane & 15, kb = (lane >> 4) * 8, ncol = lane & 15;
  const int n = wv*16 + ncol;
  v8f acc[4];
  {
    float bb = bias[n];
    #pragma unroll
    for (int mt=0; mt<4; ++mt)
      #pragma unroll
      for (int r=0;r<8;++r) acc[mt][r] = bb;
  }
  for (int kt=0; kt<4; ++kt){
    v16h bf = *(const v16h*)(Wp + ((size_t)((kt*8 + wv)*32 + lane))*16);
    #pragma unroll
    for (int mt=0; mt<4; ++mt){
      v16h af = load_a_frag(sX, LDA, mt*16 + mrow, kt*32, kb);
      acc[mt] = WMMA_F16(af, bf, acc[mt]);
    }
  }
  #pragma unroll
  for (int mt=0; mt<4; ++mt)
    #pragma unroll
    for (int r=0;r<8;++r){
      int m = mt*16 + r + ((lane>>4)<<3);
      int row = n0 + m;
      if (row < N) out[(size_t)row*128 + n] = gelu_t(acc[mt][r]);
    }
}

// ---------------------------------------------------------------- EDGE kernel (64 edges / block)
__global__ __launch_bounds__(256) void edge_kernel(
    const float* __restrict__ x, const float* __restrict__ pos,
    const int* __restrict__ src, const int* __restrict__ dst,
    const _Float16* __restrict__ W1a, const _Float16* __restrict__ W1b,
    const float* __restrict__ w1c, const float* __restrict__ b1,
    const _Float16* __restrict__ W2, const float* __restrict__ b2,
    const float* __restrict__ Wx,
    float* __restrict__ agg, float* __restrict__ dpn, int E)
{
  __shared__ _Float16 sA[64*LDA];   // xi (=x[dst]) tile, later h tile
  __shared__ _Float16 sB[64*LDA];   // xj (=x[src]) tile, later m tile (f32)
  __shared__ float sRel[64*4];      // rel.xyz, d2
  __shared__ int sSrc[64], sDst[64];

  const int tid = threadIdx.x;
  const int e0  = blockIdx.x*64;

  if (tid < 64){
    int e = e0 + tid;
    int s_ = 0, d_ = -1;
    float rx=0, ry=0, rz=0;
    if (e < E){
      s_ = src[e]; d_ = dst[e];
      rx = pos[(size_t)d_*3+0] - pos[(size_t)s_*3+0];
      ry = pos[(size_t)d_*3+1] - pos[(size_t)s_*3+1];
      rz = pos[(size_t)d_*3+2] - pos[(size_t)s_*3+2];
    }
    sSrc[tid] = s_; sDst[tid] = d_;
    sRel[tid*4+0]=rx; sRel[tid*4+1]=ry; sRel[tid*4+2]=rz;
    sRel[tid*4+3]=rx*rx + ry*ry + rz*rz;
  }
  __syncthreads();

  { // gather features -> f16 LDS tiles
    int e  = tid >> 2;
    int c0 = (tid & 3) * 32;
    int rs = sSrc[e];
    int rd = sDst[e] < 0 ? 0 : sDst[e];
    const float4* pj = (const float4*)(x + (size_t)rs*128 + c0);
    const float4* pi = (const float4*)(x + (size_t)rd*128 + c0);
    _Float16* qa = &sA[e*LDA + c0];
    _Float16* qb = &sB[e*LDA + c0];
    #pragma unroll
    for (int k=0;k<8;++k){
      float4 vi = pi[k], vj = pj[k];
      h4 hi = { (_Float16)vi.x,(_Float16)vi.y,(_Float16)vi.z,(_Float16)vi.w };
      h4 hj = { (_Float16)vj.x,(_Float16)vj.y,(_Float16)vj.z,(_Float16)vj.w };
      *(h4*)(qa + 4*k) = hi;
      *(h4*)(qb + 4*k) = hj;
    }
  }
  __syncthreads();

  const int lane = tid & 31, wv = tid >> 5;
  const int mrow = lane & 15, kb = (lane >> 4) * 8, ncol = lane & 15;

  // ----- layer 1: h = gelu(xi@W1a + xj@W1b + d2*w1c + b1); wave wv -> N-tile wv (of 8)
  v8f acc[4];
  {
    int n = wv*16 + ncol;
    float bb = b1[n], wc = w1c[n];
    #pragma unroll
    for (int mt=0; mt<4; ++mt)
      #pragma unroll
      for (int r=0;r<8;++r){
        int m = mt*16 + r + ((lane>>4)<<3);
        acc[mt][r] = bb + sRel[m*4+3]*wc;
      }
  }
  for (int kt=0; kt<4; ++kt){
    v16h bfa = *(const v16h*)(W1a + ((size_t)((kt*8 + wv)*32 + lane))*16);
    v16h bfb = *(const v16h*)(W1b + ((size_t)((kt*8 + wv)*32 + lane))*16);
    #pragma unroll
    for (int mt=0; mt<4; ++mt){
      v16h ai = load_a_frag(sA, LDA, mt*16 + mrow, kt*32, kb);
      acc[mt] = WMMA_F16(ai, bfa, acc[mt]);
      v16h aj = load_a_frag(sB, LDA, mt*16 + mrow, kt*32, kb);
      acc[mt] = WMMA_F16(aj, bfb, acc[mt]);
    }
  }
  __syncthreads();               // all waves done reading sA/sB
  {
    int n = wv*16 + ncol;
    #pragma unroll
    for (int mt=0; mt<4; ++mt)
      #pragma unroll
      for (int r=0;r<8;++r){
        int m = mt*16 + r + ((lane>>4)<<3);
        sA[m*LDA + n] = (_Float16)gelu_t(acc[mt][r]);
      }
  }
  __syncthreads();

  // ----- layer 2: m = gelu(h@W2 + b2); 16 tiles over 8 waves (2 each)
  float* sMf = (float*)sB;       // 64 x LDM f32
  {
    const int nt2 = wv & 3;
    const int mb  = (wv >> 2) * 2;
    const int n2  = nt2*16 + ncol;
    v8f a2[2];
    {
      float bb2 = b2[n2];
      #pragma unroll
      for (int j=0;j<2;++j)
        #pragma unroll
        for (int r=0;r<8;++r) a2[j][r] = bb2;
    }
    for (int kt=0; kt<4; ++kt){
      v16h bf = *(const v16h*)(W2 + ((size_t)((kt*4 + nt2)*32 + lane))*16);
      #pragma unroll
      for (int j=0;j<2;++j){
        v16h af = load_a_frag(sA, LDA, (mb+j)*16 + mrow, kt*32, kb);
        a2[j] = WMMA_F16(af, bf, a2[j]);
      }
    }
    #pragma unroll
    for (int j=0;j<2;++j)
      #pragma unroll
      for (int r=0;r<8;++r){
        int m = (mb+j)*16 + r + ((lane>>4)<<3);
        sMf[m*LDM + n2] = gelu_t(a2[j][r]);
      }
  }
  __syncthreads();

  // ----- scatter: agg[dst] += m ; dpn[dst] += rel * (m@Wx)
  {
    int c = tid & 63;
    for (int e = tid >> 6; e < 64; e += 4){
      int d = sDst[e];
      if (d >= 0) atom_add(&agg[(size_t)d*64 + c], sMf[e*LDM + c]);
    }
  }
  if (tid < 64){
    int d = sDst[tid];
    if (d >= 0){
      float w = 0.f;
      for (int c=0;c<64;++c) w += sMf[tid*LDM + c]*Wx[c];
      atom_add(&dpn[(size_t)d*3+0], sRel[tid*4+0]*w);
      atom_add(&dpn[(size_t)d*3+1], sRel[tid*4+1]*w);
      atom_add(&dpn[(size_t)d*3+2], sRel[tid*4+2]*w);
    }
  }
}

// ---------------------------------------------------------------- NODE kernel (64 nodes / block)
__global__ __launch_bounds__(256) void node_kernel(
    const float* __restrict__ x, const float* __restrict__ agg,
    const float* __restrict__ dpn, const float* __restrict__ cnt,
    const float* __restrict__ pos_in,
    const _Float16* __restrict__ Wn1p, const float* __restrict__ bn1,
    const _Float16* __restrict__ Wn2p, const float* __restrict__ bn2,
    float* __restrict__ zout, float* __restrict__ pout, int N)
{
  __shared__ _Float16 sX[64*LDA];   // x tile, later h2 tile
  __shared__ _Float16 sG[64*LDG];   // agg tile
  const int tid = threadIdx.x;
  const int n0  = blockIdx.x*64;
  {
    int r = tid >> 2;
    int row = n0 + r; if (row >= N) row = N-1;
    int c0 = (tid & 3) * 32;
    const float4* px = (const float4*)(x + (size_t)row*128 + c0);
    _Float16* q = &sX[r*LDA + c0];
    #pragma unroll
    for (int k=0;k<8;++k){
      float4 v = px[k];
      h4 h = { (_Float16)v.x,(_Float16)v.y,(_Float16)v.z,(_Float16)v.w };
      *(h4*)(q + 4*k) = h;
    }
    int cg = (tid & 3) * 16;
    const float4* pg = (const float4*)(agg + (size_t)row*64 + cg);
    _Float16* qg = &sG[r*LDG + cg];
    #pragma unroll
    for (int k=0;k<4;++k){
      float4 v = pg[k];
      h4 h = { (_Float16)v.x,(_Float16)v.y,(_Float16)v.z,(_Float16)v.w };
      *(h4*)(qg + 4*k) = h;
    }
  }
  __syncthreads();

  const int lane = tid & 31, wv = tid >> 5;
  const int mrow = lane & 15, kb = (lane >> 4) * 8, ncol = lane & 15;
  const int n = wv*16 + ncol;

  // layer 1: h2 = gelu(cat(x,agg)@Wn1 + bn1), K = 192
  v8f acc[4];
  {
    float bb = bn1[n];
    #pragma unroll
    for (int mt=0; mt<4; ++mt)
      #pragma unroll
      for (int r=0;r<8;++r) acc[mt][r] = bb;
  }
  for (int kt=0; kt<6; ++kt){
    v16h bf = *(const v16h*)(Wn1p + ((size_t)((kt*8 + wv)*32 + lane))*16);
    #pragma unroll
    for (int mt=0; mt<4; ++mt){
      v16h af = (kt < 4) ? load_a_frag(sX, LDA, mt*16 + mrow, kt*32, kb)
                         : load_a_frag(sG, LDG, mt*16 + mrow, (kt-4)*32, kb);
      acc[mt] = WMMA_F16(af, bf, acc[mt]);
    }
  }
  __syncthreads();
  #pragma unroll
  for (int mt=0; mt<4; ++mt)
    #pragma unroll
    for (int r=0;r<8;++r){
      int m = mt*16 + r + ((lane>>4)<<3);
      sX[m*LDA + n] = (_Float16)gelu_t(acc[mt][r]);
    }
  __syncthreads();

  // layer 2: out = x + h2@Wn2 + bn2
  v8f a2[4];
  {
    float bb = bn2[n];
    #pragma unroll
    for (int mt=0; mt<4; ++mt)
      #pragma unroll
      for (int r=0;r<8;++r) a2[mt][r] = bb;
  }
  for (int kt=0; kt<4; ++kt){
    v16h bf = *(const v16h*)(Wn2p + ((size_t)((kt*8 + wv)*32 + lane))*16);
    #pragma unroll
    for (int mt=0; mt<4; ++mt){
      v16h af = load_a_frag(sX, LDA, mt*16 + mrow, kt*32, kb);
      a2[mt] = WMMA_F16(af, bf, a2[mt]);
    }
  }
  #pragma unroll
  for (int mt=0; mt<4; ++mt)
    #pragma unroll
    for (int r=0;r<8;++r){
      int m = mt*16 + r + ((lane>>4)<<3);
      int row = n0 + m;
      if (row < N) zout[(size_t)row*128 + n] = x[(size_t)row*128 + n] + a2[mt][r];
    }
  if (tid < 64){
    int row = n0 + tid;
    if (row < N){
      float c = fmaxf(cnt[row], 1.0f);
      #pragma unroll
      for (int k=0;k<3;++k)
        pout[(size_t)row*3+k] = pos_in[(size_t)row*3+k] + dpn[(size_t)row*3+k]/c;
    }
  }
}

// ================================================================ host
struct ChanRaw {
  const float *We1, *be1, *We2, *be2, *Wx, *Wn1, *bn1, *Wn2, *bn2;
};
struct ChanPacked { _Float16 *W1a, *W1b, *W2, *Wn1, *Wn2; };

static void fill_chan(ChanRaw& c, void* const* d_in, const int* sz, int b){
  if (sz[b+1] == 128*64){ // sorted-key order: We1,We2,Wn1,Wn2,Wx,be1,be2,bn1,bn2
    c.We1=(const float*)d_in[b+0]; c.We2=(const float*)d_in[b+1];
    c.Wn1=(const float*)d_in[b+2]; c.Wn2=(const float*)d_in[b+3];
    c.Wx =(const float*)d_in[b+4]; c.be1=(const float*)d_in[b+5];
    c.be2=(const float*)d_in[b+6]; c.bn1=(const float*)d_in[b+7];
    c.bn2=(const float*)d_in[b+8];
  } else {                // insertion order: We1,be1,We2,be2,Wx,Wn1,bn1,Wn2,bn2
    c.We1=(const float*)d_in[b+0]; c.be1=(const float*)d_in[b+1];
    c.We2=(const float*)d_in[b+2]; c.be2=(const float*)d_in[b+3];
    c.Wx =(const float*)d_in[b+4]; c.Wn1=(const float*)d_in[b+5];
    c.bn1=(const float*)d_in[b+6]; c.Wn2=(const float*)d_in[b+7];
    c.bn2=(const float*)d_in[b+8];
  }
}

extern "C" void kernel_launch(void* const* d_in, const int* in_sizes, int n_in,
                              void* d_out, int out_size, void* d_ws, size_t ws_size,
                              hipStream_t stream)
{
  (void)n_in; (void)out_size; (void)ws_size;
  const int N = in_sizes[0] / 128;
  const int E = in_sizes[2] / 2;
  const float* x   = (const float*)d_in[0];
  const float* pos = (const float*)d_in[1];
  const int*   ei  = (const int*)d_in[2];
  const int* src = ei;
  const int* dst = ei + E;

  const float *injW, *injB;
  ChanRaw c0, c1;
  if (in_sizes[3] == 128*128 && in_sizes[4] == 128){ // insertion order: inject first
    injW = (const float*)d_in[3]; injB = (const float*)d_in[4];
    fill_chan(c0, d_in, in_sizes, 5);
    fill_chan(c1, d_in, in_sizes, 14);
  } else {                                           // sorted: ch0, ch1, inject
    fill_chan(c0, d_in, in_sizes, 3);
    fill_chan(c1, d_in, in_sizes, 12);
    injW = (const float*)d_in[21]; injB = (const float*)d_in[22];
  }

  // ---- workspace carve-up (floats)
  float* cur = (float*)d_ws;
  auto alloc = [&](size_t nf){ float* p = cur; cur += nf; return p; };
  float* z_carry = alloc((size_t)N*128);
  float* p_carry = alloc((size_t)N*3);
  float* xinj    = alloc((size_t)N*128);
  float* z0      = alloc((size_t)N*128);
  float* gz1     = alloc((size_t)N*128);
  float* z2      = alloc((size_t)N*128);
  float* p0b     = alloc((size_t)N*3);
  float* p1b     = alloc((size_t)N*3);
  float* p2b     = alloc((size_t)N*3);
  float* aggb    = alloc((size_t)N*64);
  float* dpnb    = alloc((size_t)N*3);
  float* cntb    = alloc((size_t)N);
  float* stats   = alloc(8);
  float* partial = alloc(128*6);
  auto allocH = [&](size_t nh){ _Float16* p = (_Float16*)cur; cur += (nh+1)/2; return p; };
  _Float16* pInjW = allocH(16384);
  ChanPacked pk0, pk1;
  pk0.W1a = allocH(16384); pk0.W1b = allocH(16384); pk0.W2 = allocH(8192);
  pk0.Wn1 = allocH(24576); pk0.Wn2 = allocH(16384);
  pk1.W1a = allocH(16384); pk1.W1b = allocH(16384); pk1.W2 = allocH(8192);
  pk1.Wn1 = allocH(24576); pk1.Wn2 = allocH(16384);

  auto zero = [&](float* p, long n){
    int blocks = (int)((n + 255) / 256); if (blocks > 4096) blocks = 4096;
    zero_kernel<<<blocks, 256, 0, stream>>>(p, n);
  };
  auto pack = [&](const float* W, int K, int Nc, int ld, _Float16* out){
    int total = (K/32)*(Nc/16)*512;
    pack_b_kernel<<<(total+255)/256, 256, 0, stream>>>(W, K, Nc, ld, out);
  };

  // ---- prep: counts + packed weights + x_inj
  zero(cntb, N);
  count_kernel<<<(E+255)/256, 256, 0, stream>>>(dst, cntb, E);
  pack(injW, 128,128,128, pInjW);
  pack(c0.We1,           128,128,128, pk0.W1a);
  pack(c0.We1 + 128*128, 128,128,128, pk0.W1b);
  pack(c0.We2,           128, 64, 64, pk0.W2);
  pack(c0.Wn1,           192,128,128, pk0.Wn1);
  pack(c0.Wn2,           128,128,128, pk0.Wn2);
  pack(c1.We1,           128,128,128, pk1.W1a);
  pack(c1.We1 + 128*128, 128,128,128, pk1.W1b);
  pack(c1.We2,           128, 64, 64, pk1.W2);
  pack(c1.Wn1,           192,128,128, pk1.Wn1);
  pack(c1.Wn2,           128,128,128, pk1.Wn2);

  const int lnGrid   = (N + 7) / 8;
  const int tileGrid = (N + 63) / 64;
  // x_inj = LN(gelu(LN(x)@W + b))   (use z0/gz1 as temps)
  ln_kernel<<<lnGrid, 256, 0, stream>>>(x, nullptr, z0, N, 0);
  gemm128_gelu_kernel<<<tileGrid, 256, 0, stream>>>(z0, pInjW, injB, gz1, N);
  ln_kernel<<<lnGrid, 256, 0, stream>>>(gz1, nullptr, xinj, N, 0);

  zero(z_carry, (long)N*128);
  zero(p_carry, (long)N*3);

  auto run_channel = [&](const ChanRaw& cr, const ChanPacked& cp,
                         const float* xin, const float* posin,
                         float* zo, float* po){
    zero(aggb, (long)N*64);
    zero(dpnb, (long)N*3);
    edge_kernel<<<(E+63)/64, 256, 0, stream>>>(xin, posin, src, dst,
        cp.W1a, cp.W1b, cr.We1 + 256*128, cr.be1, cp.W2, cr.be2, cr.Wx,
        aggb, dpnb, E);
    node_kernel<<<tileGrid, 256, 0, stream>>>(xin, aggb, dpnb, cntb, posin,
        cp.Wn1, cr.bn1, cp.Wn2, cr.bn2, zo, po, N);
  };
  auto bn = [&](const float* a, const float* b, float* out){
    bn_reduce_kernel<<<128, 256, 0, stream>>>(a, b, partial, N);
    bn_final_kernel<<<1, 32, 0, stream>>>(partial, 128, stats);
    bn_apply_kernel<<<(N*3 + 255)/256, 256, 0, stream>>>(a, b, stats, out, N);
  };

  for (int it = 0; it < 12; ++it){
    // channel 0: uses carry z and ORIGINAL pos for rel
    run_channel(c0, pk0, z_carry, pos, z0, p0b);
    ln_kernel<<<lnGrid, 256, 0, stream>>>(z0, xinj, gz1, N, 1);   // gz1 = gelu(LN(z0+xinj))
    bn(p0b, p_carry, p1b);                                        // p1 = BN(p0 + p)
    // channel 1: uses gz1 and p1
    run_channel(c1, pk1, gz1, p1b, z2, p2b);
    float* zo = (it == 11) ? (float*)d_out : z_carry;
    float* po = (it == 11) ? ((float*)d_out + (size_t)N*128) : p_carry;
    ln_kernel<<<lnGrid, 256, 0, stream>>>(z0, z2, zo, N, 2);      // z3 = LN(gelu(z0+z2))
    bn(p0b, p2b, po);                                             // p3 = BN(p0 + p2)
  }
}
